// DNC_51513837748319
// MI455X (gfx1250) — matrix operand order repaired and checked by reading, not compiled, not run on hardware
//
#include <hip/hip_runtime.h>

// ---------------------------------------------------------------------------
// DNC forward on gfx1250: bf16 WMMA GEMMs + fused LN / LSTM / memory kernels.
// B=64, L=256, E=512, H=512, N=128, D=64, R=4, O=2048, CI=768, TO=768
// ---------------------------------------------------------------------------

typedef __attribute__((ext_vector_type(16))) __bf16 bf16x16;
typedef __attribute__((ext_vector_type(8)))  float  f32x8;
typedef __attribute__((ext_vector_type(8)))  unsigned int u32x8;

#define EPSLN 1e-5f

__device__ __forceinline__ unsigned short f32_to_bf16(float f) {
    unsigned int u = __float_as_uint(f);
    u += 0x7fffu + ((u >> 16) & 1u);   // round-to-nearest-even
    return (unsigned short)(u >> 16);
}

__device__ __forceinline__ float sigmf(float x) { return 1.0f / (1.0f + __expf(-x)); }

__device__ __forceinline__ bf16x16 as_bf16x16(u32x8 u) {
    union { u32x8 u; bf16x16 h; } c; c.u = u; return c.h;
}

// ---------------------------------------------------------------------------
// WMMA GEMM, M fixed = 64.  C[64 x N] = A[64 x K](bf16) * Bw[N x K]^T(bf16) + bias
// Each wave computes a 16x64 output block (4 accumulators, A reused 4x).
// Launch: grid = (4 * N/64) / 8 blocks of 256 threads.
// ---------------------------------------------------------------------------
__device__ __forceinline__ void store_acc(float* __restrict__ C, long ldc,
                                          int mb, int nc, float bb, f32x8 acc) {
#pragma unroll
    for (int r = 0; r < 8; ++r) C[(long)(mb + r) * ldc + nc] = acc[r] + bb;
}

__global__ void __launch_bounds__(256)
wmma_gemm_m64(const unsigned short* __restrict__ A,
              const unsigned short* __restrict__ Bw,
              const float* __restrict__ bias,
              float* __restrict__ C, long ldc, int N, int K) {
    const int lane = threadIdx.x & 31;
    const int wid  = (blockIdx.x * blockDim.x + threadIdx.x) >> 5;
    const int colGroups = N >> 6;
    const int rowTile  = wid / colGroups;
    const int colGroup = wid - rowTile * colGroups;
    const int m0 = rowTile << 4;
    const int n0 = colGroup << 6;
    const int half = lane >> 4;      // 0: lanes 0-15, 1: lanes 16-31
    const int l15  = lane & 15;

    const unsigned short* arow = A + (long)(m0 + l15) * K;
    const unsigned short* bp0 = Bw + (long)(n0 +  0 + l15) * K + (half << 4);
    const unsigned short* bp1 = Bw + (long)(n0 + 16 + l15) * K + (half << 4);
    const unsigned short* bp2 = Bw + (long)(n0 + 32 + l15) * K + (half << 4);
    const unsigned short* bp3 = Bw + (long)(n0 + 48 + l15) * K + (half << 4);

    f32x8 acc0 = {}, acc1 = {}, acc2 = {}, acc3 = {};

    for (int k0 = 0; k0 < K; k0 += 32) {
        // A tile 16x32: lane<16 holds K {0..15}, lane>=16 holds {16..31}
        const uint4 alo = *(const uint4*)(arow + k0 + (half << 4));
        const uint4 ahi = *(const uint4*)(arow + k0 + 8 + (half << 4));
        u32x8 au;
        au[0] = alo.x; au[1] = alo.y; au[2] = alo.z; au[3] = alo.w;
        au[4] = ahi.x; au[5] = ahi.y; au[6] = ahi.z; au[7] = ahi.w;
        const bf16x16 a = as_bf16x16(au);
        // B tile 32x16 per col group: lane = column, 16 contiguous K from weight row
        const bf16x16 b0 = as_bf16x16(*(const u32x8*)(bp0 + k0));
        const bf16x16 b1 = as_bf16x16(*(const u32x8*)(bp1 + k0));
        const bf16x16 b2 = as_bf16x16(*(const u32x8*)(bp2 + k0));
        const bf16x16 b3 = as_bf16x16(*(const u32x8*)(bp3 + k0));
        acc0 = __builtin_amdgcn_wmma_f32_16x16x32_bf16(false, a, false, b0, (short)0, acc0, false, false);
        acc1 = __builtin_amdgcn_wmma_f32_16x16x32_bf16(false, a, false, b1, (short)0, acc1, false, false);
        acc2 = __builtin_amdgcn_wmma_f32_16x16x32_bf16(false, a, false, b2, (short)0, acc2, false, false);
        acc3 = __builtin_amdgcn_wmma_f32_16x16x32_bf16(false, a, false, b3, (short)0, acc3, false, false);
    }

    const int mb = m0 + (half << 3);
    {
        int nc = n0 + l15;
        store_acc(C, ldc, mb, nc, bias ? bias[nc] : 0.0f, acc0);
        nc += 16; store_acc(C, ldc, mb, nc, bias ? bias[nc] : 0.0f, acc1);
        nc += 16; store_acc(C, ldc, mb, nc, bias ? bias[nc] : 0.0f, acc2);
        nc += 16; store_acc(C, ldc, mb, nc, bias ? bias[nc] : 0.0f, acc3);
    }
}

// ---------------------------------------------------------------------------
// Step kernel 1: ci = LN(concat(x_t, rv)) -> bf16 into A_g[:, 0:768]
// one block (256 thr) per batch row, 768 elements (3 per thread)
// ---------------------------------------------------------------------------
__global__ void __launch_bounds__(256)
k_ci(const float* __restrict__ x_emb, const float* __restrict__ rv,
     const float* __restrict__ g_in, const float* __restrict__ b_in,
     unsigned short* __restrict__ Ag, int t) {
    __shared__ float2 red[256];
    const int b = blockIdx.x, tid = threadIdx.x;
    float v[3]; float s = 0.f, q = 0.f;
#pragma unroll
    for (int i = 0; i < 3; ++i) {
        const int j = tid + (i << 8);
        const float x = (j < 512) ? x_emb[((long)b * 256 + t) * 512 + j]
                                  : rv[b * 256 + (j - 512)];
        v[i] = x; s += x; q += x * x;
    }
    red[tid] = make_float2(s, q);
    for (int off = 128; off > 0; off >>= 1) {
        __syncthreads();
        if (tid < off) { red[tid].x += red[tid + off].x; red[tid].y += red[tid + off].y; }
    }
    __syncthreads();
    const float mean = red[0].x * (1.0f / 768.0f);
    const float var  = red[0].y * (1.0f / 768.0f) - mean * mean;
    const float rstd = rsqrtf(var + EPSLN);
#pragma unroll
    for (int i = 0; i < 3; ++i) {
        const int j = tid + (i << 8);
        Ag[b * 1280 + j] = f32_to_bf16((v[i] - mean) * rstd * g_in[j] + b_in[j]);
    }
}

// ---------------------------------------------------------------------------
// Step kernel 3: LSTM cell + hn = LN(h).  h(bf16)->A_g[:,768:1280], hn(bf16)->A_h
// ---------------------------------------------------------------------------
__global__ void __launch_bounds__(256)
k_lstm(const float* __restrict__ gates, float* __restrict__ hbuf, float* __restrict__ cbuf,
       const float* __restrict__ g_ctrl, const float* __restrict__ b_ctrl,
       float* __restrict__ hn, unsigned short* __restrict__ Ag,
       unsigned short* __restrict__ Ah) {
    __shared__ float2 red[256];
    const int b = blockIdx.x, tid = threadIdx.x;
    const float* g = gates + (long)b * 2048;
    float hv[2]; float s = 0.f, q = 0.f;
#pragma unroll
    for (int i = 0; i < 2; ++i) {
        const int e = tid + (i << 8);
        const float ig = sigmf(g[e]);
        const float fg = sigmf(g[512 + e]);
        const float gg = tanhf(g[1024 + e]);
        const float og = sigmf(g[1536 + e]);
        const float cn = fg * cbuf[b * 512 + e] + ig * gg;
        const float h  = og * tanhf(cn);
        cbuf[b * 512 + e] = cn;
        hbuf[b * 512 + e] = h;
        Ag[b * 1280 + 768 + e] = f32_to_bf16(h);
        hv[i] = h; s += h; q += h * h;
    }
    red[tid] = make_float2(s, q);
    for (int off = 128; off > 0; off >>= 1) {
        __syncthreads();
        if (tid < off) { red[tid].x += red[tid + off].x; red[tid].y += red[tid + off].y; }
    }
    __syncthreads();
    const float mean = red[0].x * (1.0f / 512.0f);
    const float var  = red[0].y * (1.0f / 512.0f) - mean * mean;
    const float rstd = rsqrtf(var + EPSLN);
#pragma unroll
    for (int i = 0; i < 2; ++i) {
        const int e = tid + (i << 8);
        const float y = (hv[i] - mean) * rstd * g_ctrl[e] + b_ctrl[e];
        hn[b * 512 + e] = y;
        Ah[b * 512 + e] = f32_to_bf16(y);
    }
}

// ---------------------------------------------------------------------------
// Step kernel 5: memory write + read + output LN.  One block (256 thr) per batch.
// Memory tile (128x64 fp32 = 32KB) resident in LDS for the whole step.
// ---------------------------------------------------------------------------
__global__ void __launch_bounds__(256)
k_memory(float* __restrict__ mem, const float* __restrict__ headout,
         const float* __restrict__ hn,
         const float* __restrict__ g_rk, const float* __restrict__ b_ln_rk,
         const float* __restrict__ g_wk, const float* __restrict__ b_ln_wk,
         const float* __restrict__ g_mem, const float* __restrict__ b_ln_mem,
         const float* __restrict__ g_out, const float* __restrict__ b_out,
         float* __restrict__ rv, unsigned short* __restrict__ Ap) {
    __shared__ float  memS[128 * 64];
    __shared__ float2 red2[256];
    __shared__ float  redf[256];
    __shared__ float  wkn[64], erS[64], adS[64], gm[64], bm[64];
    __shared__ float  ww[128], mu2[128], rs2[128];
    __shared__ float  rkn[256], rwS[512], rvL[256];
    __shared__ float  wsS;

    const int b = blockIdx.x, tid = threadIdx.x;
    const float* ho = headout + (long)b * 512;
    float* gmem = mem + (long)b * 128 * 64;

    for (int i = tid; i < 128 * 64; i += 256) memS[i] = gmem[i];
    if (tid < 64) {
        gm[tid]  = g_mem[tid];  bm[tid] = b_ln_mem[tid];
        erS[tid] = sigmf(ho[321 + tid]);
        adS[tid] = tanhf(ho[385 + tid]);
    }
    if (tid == 0) wsS = sigmf(ho[320]);

    // wkn = LN(wk)
    const float wraw = (tid < 64) ? ho[256 + tid] : 0.0f;
    red2[tid] = make_float2(wraw, wraw * wraw);
    for (int off = 128; off > 0; off >>= 1) {
        __syncthreads();
        if (tid < off) { red2[tid].x += red2[tid + off].x; red2[tid].y += red2[tid + off].y; }
    }
    __syncthreads();
    {
        const float mw = red2[0].x * (1.0f / 64.0f);
        const float vw = red2[0].y * (1.0f / 64.0f) - mw * mw;
        const float rw = rsqrtf(vw + EPSLN);
        if (tid < 64) wkn[tid] = (wraw - mw) * rw * g_wk[tid] + b_ln_wk[tid];
    }
    __syncthreads();

    // write scores: dot(LN(mem[n]), wkn)
    float sc = 0.0f;
    if (tid < 128) {
        const float* row = memS + tid * 64;
        float s = 0.f, q = 0.f;
        for (int d = 0; d < 64; ++d) { const float v = row[d]; s += v; q += v * v; }
        const float mu = s * (1.0f / 64.0f);
        const float rs = rsqrtf(q * (1.0f / 64.0f) - mu * mu + EPSLN);
        float dv = 0.f;
        for (int d = 0; d < 64; ++d) dv += ((row[d] - mu) * rs * gm[d] + bm[d]) * wkn[d];
        sc = dv;
    }
    // softmax over n, scaled by write strength
    redf[tid] = (tid < 128) ? sc : -3.0e38f;
    for (int off = 128; off > 0; off >>= 1) {
        __syncthreads();
        if (tid < off) redf[tid] = fmaxf(redf[tid], redf[tid + off]);
    }
    __syncthreads();
    const float mxw = redf[0];
    __syncthreads();
    const float ew = (tid < 128) ? __expf(sc - mxw) : 0.0f;
    redf[tid] = ew;
    for (int off = 128; off > 0; off >>= 1) {
        __syncthreads();
        if (tid < off) redf[tid] += redf[tid + off];
    }
    __syncthreads();
    if (tid < 128) ww[tid] = ew / redf[0] * wsS;
    __syncthreads();

    // memory update + post-write LN stats
    if (tid < 128) {
        float* row = memS + tid * 64;
        const float w = ww[tid];
        float s = 0.f, q = 0.f;
        for (int d = 0; d < 64; ++d) {
            const float nv = row[d] * (1.0f - w * erS[d]) + w * adS[d];
            row[d] = nv; s += nv; q += nv * nv;
        }
        const float mu = s * (1.0f / 64.0f);
        mu2[tid] = mu;
        rs2[tid] = rsqrtf(q * (1.0f / 64.0f) - mu * mu + EPSLN);
    }
    __syncthreads();
    for (int i = tid; i < 128 * 64; i += 256) gmem[i] = memS[i];

    // rkn = LN(rk[r]) -- segmented reductions of 64 within the 256 threads
    {
        const int r = tid >> 6, d = tid & 63;
        const float raw = ho[r * 64 + d];
        red2[tid] = make_float2(raw, raw * raw);
        for (int off = 32; off > 0; off >>= 1) {
            __syncthreads();
            if (d < off) { red2[tid].x += red2[tid + off].x; red2[tid].y += red2[tid + off].y; }
        }
        __syncthreads();
        const float2 sg = red2[tid & ~63];
        const float mu = sg.x * (1.0f / 64.0f);
        const float rs = rsqrtf(sg.y * (1.0f / 64.0f) - mu * mu + EPSLN);
        rkn[tid] = (raw - mu) * rs * g_rk[d] + b_ln_rk[d];
    }
    __syncthreads();

    // read scores rw[r][n] = dot(rkn[r], LN(mem_new[n]))
    for (int idx = tid; idx < 512; idx += 256) {
        const int r = idx >> 7, n = idx & 127;
        const float* row = memS + n * 64;
        const float* kr  = rkn + r * 64;
        const float mu = mu2[n], rs = rs2[n];
        float dv = 0.f;
        for (int d = 0; d < 64; ++d) dv += ((row[d] - mu) * rs * gm[d] + bm[d]) * kr[d];
        rwS[idx] = dv;
    }
    __syncthreads();

    // softmax over n for each read head
    for (int r = 0; r < 4; ++r) {
        const float v = (tid < 128) ? rwS[r * 128 + tid] : -3.0e38f;
        __syncthreads();
        redf[tid] = v;
        for (int off = 128; off > 0; off >>= 1) {
            __syncthreads();
            if (tid < off) redf[tid] = fmaxf(redf[tid], redf[tid + off]);
        }
        __syncthreads();
        const float mx = redf[0];
        __syncthreads();
        const float e = (tid < 128) ? __expf(v - mx) : 0.0f;
        redf[tid] = e;
        for (int off = 128; off > 0; off >>= 1) {
            __syncthreads();
            if (tid < off) redf[tid] += redf[tid + off];
        }
        __syncthreads();
        const float sm = redf[0];
        __syncthreads();
        if (tid < 128) rwS[r * 128 + tid] = e / sm;
    }
    __syncthreads();

    // read vectors rv[r][d] = sum_n rw[r][n] * mem_new[n][d]
    {
        const int r = tid >> 6, d = tid & 63;
        const float* wr = rwS + r * 128;
        float acc = 0.f;
        for (int n = 0; n < 128; ++n) acc += wr[n] * memS[n * 64 + d];
        rv[b * 256 + tid] = acc;
        rvL[tid] = acc;
    }
    __syncthreads();

    // out = LN(concat(hn, rv)) -> bf16 projection input
    float ov[3]; float s = 0.f, q = 0.f;
#pragma unroll
    for (int i = 0; i < 3; ++i) {
        const int j = tid + (i << 8);
        const float x = (j < 512) ? hn[b * 512 + j] : rvL[j - 512];
        ov[i] = x; s += x; q += x * x;
    }
    red2[tid] = make_float2(s, q);
    for (int off = 128; off > 0; off >>= 1) {
        __syncthreads();
        if (tid < off) { red2[tid].x += red2[tid + off].x; red2[tid].y += red2[tid + off].y; }
    }
    __syncthreads();
    const float mean = red2[0].x * (1.0f / 768.0f);
    const float var  = red2[0].y * (1.0f / 768.0f) - mean * mean;
    const float rstd = rsqrtf(var + EPSLN);
#pragma unroll
    for (int i = 0; i < 3; ++i) {
        const int j = tid + (i << 8);
        Ap[b * 768 + j] = f32_to_bf16((ov[i] - mean) * rstd * g_out[j] + b_out[j]);
    }
}

// ---------------------------------------------------------------------------
// Preprocessing / init / finalize
// ---------------------------------------------------------------------------
__global__ void k_zero(unsigned int* __restrict__ p, long n) {
    long i = (long)blockIdx.x * blockDim.x + threadIdx.x;
    const long stride = (long)gridDim.x * blockDim.x;
    for (; i < n; i += stride) p[i] = 0u;
}

__global__ void k_cvt(const float* __restrict__ src, unsigned short* __restrict__ dst,
                      int rows, int cols, int dstLd, int dstRowOff, int dstColOff) {
    long i = (long)blockIdx.x * blockDim.x + threadIdx.x;
    const long total = (long)rows * cols;
    const long stride = (long)gridDim.x * blockDim.x;
    for (; i < total; i += stride) {
        const int r = (int)(i / cols), c = (int)(i - (long)r * cols);
        dst[(long)(r + dstRowOff) * dstLd + dstColOff + c] = f32_to_bf16(src[i]);
    }
}

__global__ void k_bias(const float* __restrict__ b_ih, const float* __restrict__ b_hh,
                       float* __restrict__ bias_g,
                       const float* __restrict__ b_rk, const float* __restrict__ b_wk,
                       const float* __restrict__ b_ws, const float* __restrict__ b_er,
                       const float* __restrict__ b_ad, float* __restrict__ bias_h) {
    const int i = blockIdx.x * blockDim.x + threadIdx.x;
    if (i < 2048) bias_g[i] = b_ih[i] + b_hh[i];
    if (i < 512) {
        float v = 0.0f;
        if (i < 256)      v = b_rk[i];
        else if (i < 320) v = b_wk[i - 256];
        else if (i == 320) v = b_ws[0];
        else if (i < 385) v = b_er[i - 321];
        else if (i < 449) v = b_ad[i - 385];
        bias_h[i] = v;
    }
}

__global__ void k_finalize(const float* __restrict__ mem, const float* __restrict__ h,
                           const float* __restrict__ c, float* __restrict__ dst) {
    const long i = (long)blockIdx.x * blockDim.x + threadIdx.x;
    const long M = 64L * 128 * 64;          // 524288
    if (i < M)                dst[i] = mem[i];
    else if (i < M + 32768)   dst[i] = h[i - M];
    else if (i < M + 65536)   dst[i] = c[i - M - 32768];
}

// ---------------------------------------------------------------------------
// Host driver
// ---------------------------------------------------------------------------
extern "C" void kernel_launch(void* const* d_in, const int* in_sizes, int n_in,
                              void* d_out, int out_size, void* d_ws, size_t ws_size,
                              hipStream_t stream) {
    (void)in_sizes; (void)n_in; (void)out_size; (void)ws_size;

    const float* x_emb   = (const float*)d_in[0];
    const float* g_in    = (const float*)d_in[1];
    const float* b_in    = (const float*)d_in[2];
    const float* W_ih    = (const float*)d_in[3];
    const float* W_hh    = (const float*)d_in[4];
    const float* b_ih    = (const float*)d_in[5];
    const float* b_hh    = (const float*)d_in[6];
    const float* g_ctrl  = (const float*)d_in[7];
    const float* b_ctrl  = (const float*)d_in[8];
    const float* W_rk    = (const float*)d_in[9];
    const float* b_rk    = (const float*)d_in[10];
    const float* W_wk    = (const float*)d_in[11];
    const float* b_wk    = (const float*)d_in[12];
    const float* W_ws    = (const float*)d_in[13];
    const float* b_ws    = (const float*)d_in[14];
    const float* W_er    = (const float*)d_in[15];
    const float* b_er    = (const float*)d_in[16];
    const float* W_ad    = (const float*)d_in[17];
    const float* b_ad    = (const float*)d_in[18];
    const float* g_rk    = (const float*)d_in[19];
    const float* b_ln_rk = (const float*)d_in[20];
    const float* g_wk    = (const float*)d_in[21];
    const float* b_ln_wk = (const float*)d_in[22];
    const float* g_mem   = (const float*)d_in[23];
    const float* b_ln_mem= (const float*)d_in[24];
    const float* g_out   = (const float*)d_in[25];
    const float* b_out   = (const float*)d_in[26];
    const float* W_proj  = (const float*)d_in[27];
    const float* b_proj  = (const float*)d_in[28];

    char* ws = (char*)d_ws;
    size_t off = 0;
    auto take = [&](size_t bytes) -> char* {
        off = (off + 255) & ~(size_t)255;
        char* p = ws + off; off += bytes; return p;
    };

    unsigned short* Wg   = (unsigned short*)take(2048L * 1280 * 2); // [W_ih | W_hh] bf16
    unsigned short* Wh   = (unsigned short*)take(512L * 512 * 2);   // packed heads (padded)
    unsigned short* Wp   = (unsigned short*)take(2048L * 768 * 2);  // W_proj bf16
    float* bias_g        = (float*)take(2048 * 4);
    float* bias_h        = (float*)take(512 * 4);
    unsigned short* Ag   = (unsigned short*)take(64L * 1280 * 2);   // [ci | h] bf16
    unsigned short* Ah   = (unsigned short*)take(64L * 512 * 2);    // hn bf16
    unsigned short* Ap   = (unsigned short*)take(64L * 768 * 2);    // out-LN bf16
    float* gates         = (float*)take(64L * 2048 * 4);
    float* headout       = (float*)take(64L * 512 * 4);
    float* hbuf          = (float*)take(64L * 512 * 4);
    float* cbuf          = (float*)take(64L * 512 * 4);
    float* hnbuf         = (float*)take(64L * 512 * 4);
    float* rvbuf         = (float*)take(64L * 256 * 4);
    float* membuf        = (float*)take(64L * 128 * 64 * 4);

    // ---- one-time (per call) preprocessing ----
    k_zero<<<512, 256, 0, stream>>>((unsigned int*)Wh,     512L * 512 / 2);
    k_zero<<<512, 256, 0, stream>>>((unsigned int*)Ag,     64L * 1280 / 2);
    k_zero<<<512, 256, 0, stream>>>((unsigned int*)hbuf,   64L * 512);
    k_zero<<<512, 256, 0, stream>>>((unsigned int*)cbuf,   64L * 512);
    k_zero<<<512, 256, 0, stream>>>((unsigned int*)rvbuf,  64L * 256);
    k_zero<<<512, 256, 0, stream>>>((unsigned int*)membuf, 64L * 128 * 64);

    k_cvt<<<1024, 256, 0, stream>>>(W_ih,   Wg, 2048, 768, 1280, 0,   0);
    k_cvt<<<1024, 256, 0, stream>>>(W_hh,   Wg, 2048, 512, 1280, 0,   768);
    k_cvt<<<512,  256, 0, stream>>>(W_rk,   Wh, 256,  512, 512,  0,   0);
    k_cvt<<<256,  256, 0, stream>>>(W_wk,   Wh, 64,   512, 512,  256, 0);
    k_cvt<<<8,    256, 0, stream>>>(W_ws,   Wh, 1,    512, 512,  320, 0);
    k_cvt<<<256,  256, 0, stream>>>(W_er,   Wh, 64,   512, 512,  321, 0);
    k_cvt<<<256,  256, 0, stream>>>(W_ad,   Wh, 64,   512, 512,  385, 0);
    k_cvt<<<1024, 256, 0, stream>>>(W_proj, Wp, 2048, 768, 768,  0,   0);
    k_bias<<<8, 256, 0, stream>>>(b_ih, b_hh, bias_g, b_rk, b_wk, b_ws, b_er, b_ad, bias_h);

    // ---- recurrent scan ----
    float* out = (float*)d_out;
    for (int t = 0; t < 256; ++t) {
        k_ci<<<64, 256, 0, stream>>>(x_emb, rvbuf, g_in, b_in, Ag, t);
        wmma_gemm_m64<<<16, 256, 0, stream>>>(Ag, Wg, bias_g, gates, 2048L, 2048, 1280);
        k_lstm<<<64, 256, 0, stream>>>(gates, hbuf, cbuf, g_ctrl, b_ctrl, hnbuf, Ag, Ah);
        wmma_gemm_m64<<<4, 256, 0, stream>>>(Ah, Wh, bias_h, headout, 512L, 512, 512);
        k_memory<<<64, 256, 0, stream>>>(membuf, headout, hnbuf,
                                         g_rk, b_ln_rk, g_wk, b_ln_wk,
                                         g_mem, b_ln_mem, g_out, b_out,
                                         rvbuf, Ap);
        wmma_gemm_m64<<<16, 256, 0, stream>>>(Ap, Wp, b_proj,
                                              out + (long)t * 2048, 524288L /* L*O */,
                                              2048, 768);
    }

    // ---- final state outputs: mem, h[None], c[None] ----
    k_finalize<<<2304, 256, 0, stream>>>(membuf, hbuf, cbuf, out + 33554432L);
}
